// Searching_model_54185307406552
// MI455X (gfx1250) — compile-verified, matrix operand order checked by reading, not run on hardware
//
#include <hip/hip_runtime.h>

typedef __attribute__((ext_vector_type(16))) _Float16 v16h;
typedef __attribute__((ext_vector_type(8)))  float    v8f;

#define BDIM 64
#define LDIM 128
#define TILE 16

// One block computes a 16x16 tile of the [64,64] output.
// Phase 1: stage zh rows [i0,i0+16) and it rows [j0,j0+16) in LDS.
// Two passes over j-groups of 8:
//   Phase 2: indicator f16 (1.0 iff no exact match of zh[i][l] in it row j),
//            scattered into LDS in the EXACT WMMA A-fragment layout.
//   Phase 3: each wave reduces one j-column over l with 4x
//            v_wmma_f32_16x16x32_f16 against an all-ones B (exact arithmetic).
__global__ __launch_bounds__(256) void searching_model_kernel(
    const float* __restrict__ it_cap,   // [64,128]
    const float* __restrict__ zh_cap,   // [64,128]
    float* __restrict__ out)            // [64,64]
{
    __shared__ __align__(16) float    zh_s[TILE * LDIM];        // 8 KB
    __shared__ __align__(16) float    it_s[TILE * LDIM];        // 8 KB
    __shared__ __align__(32) _Float16 ind_h[8 * 4 * 32 * 16];   // 32 KB, A-fragment layout

    const int t  = threadIdx.x;
    const int i0 = blockIdx.y * TILE;
    const int j0 = blockIdx.x * TILE;

    // ---- Phase 1: tile loads (tiles are contiguous rows: 2048 floats each) ----
    const float4* zg = (const float4*)(zh_cap + i0 * LDIM);
    const float4* ig = (const float4*)(it_cap + j0 * LDIM);
    __builtin_prefetch(zg, 0, 0);
    __builtin_prefetch(ig, 0, 0);
    float4* zs  = (float4*)zh_s;
    float4* is4 = (float4*)it_s;
    for (int idx = t; idx < (TILE * LDIM) / 4; idx += 256) {
        zs[idx]  = zg[idx];
        is4[idx] = ig[idx];
    }
    __syncthreads();

    const int lane = t & 31;
    const int wave = t >> 5;
    const int mrow = lane & 15;
    const int half = lane >> 4;

    v16h ones;
    #pragma unroll
    for (int e = 0; e < 16; ++e) ones[e] = (_Float16)1.0f;

    // Phase-2 work mapping: thread -> (i, j_local, l-half)
    const int pi  = t >> 4;          // i in tile: 0..15
    const int pj  = (t >> 1) & 7;    // j within group: 0..7
    const int pl0 = (t & 1) * 64;    // l range start

    for (int p = 0; p < 2; ++p) {
        // ---- Phase 2: indicator f16, written directly in A-fragment layout ----
        {
            const int j = p * 8 + pj;
            const float4* itr = (const float4*)&it_s[j * LDIM];
            for (int l = pl0; l < pl0 + 64; ++l) {
                const float zv = zh_s[pi * LDIM + l];
                bool eq = false;
                #pragma unroll 8
                for (int kk = 0; kk < LDIM / 4; ++kk) {
                    const float4 v = itr[kk];
                    eq = eq || (zv == v.x) || (zv == v.y) || (zv == v.z) || (zv == v.w);
                }
                // A-layout inverse map: element (m=pi, K=l&31) of chunk l>>5
                const int k     = l & 31;
                const int chunk = l >> 5;
                const int alane = pi + (((k >> 3) & 1) << 4);
                const int e     = ((k & 16) >> 1) + (k & 7);
                ind_h[(((pj << 2) + chunk) * 32 + alane) * 16 + e] =
                    eq ? (_Float16)0.0f : (_Float16)1.0f;
            }
        }
        __syncthreads();

        // ---- Phase 3: wave w reduces column j = p*8 + w via WMMA ----
        v8f acc = {};
        #pragma unroll
        for (int chunk = 0; chunk < 4; ++chunk) {
            // 16 contiguous halves per lane, 32B-aligned -> 2x ds_load_b128
            const v16h a = *(const v16h*)&ind_h[(((wave << 2) + chunk) * 32 + lane) * 16];
            // D = A x Ones + C : every column of D = row sums of A (exact)
            acc = __builtin_amdgcn_wmma_f32_16x16x32_f16(
                false, a, false, ones, (short)0, acc, false, false);
        }
        // C/D layout: element r, lanes 0-15 -> (M=r, N=lane); lanes 16-31 -> (M=r+8).
        if (mrow == 0) {
            const int mbase = half << 3;
            #pragma unroll
            for (int r = 0; r < 8; ++r) {
                out[(i0 + mbase + r) * BDIM + (j0 + p * 8 + wave)] = acc[r];
            }
        }
        __syncthreads();  // ind_h reused next pass
    }
}

extern "C" void kernel_launch(void* const* d_in, const int* in_sizes, int n_in,
                              void* d_out, int out_size, void* d_ws, size_t ws_size,
                              hipStream_t stream) {
    // setup_inputs() order: {"it_cap", "zh_cap"} -> d_in[0]=it_cap, d_in[1]=zh_cap
    const float* it_cap = (const float*)d_in[0];
    const float* zh_cap = (const float*)d_in[1];
    float* out = (float*)d_out;

    dim3 grid(BDIM / TILE, BDIM / TILE);   // (4,4): 16 tiles of 16x16
    searching_model_kernel<<<grid, dim3(256), 0, stream>>>(it_cap, zh_cap, out);
}
